// XYZEdgeConv_43542378447070
// MI455X (gfx1250) — compile-verified, matrix operand order checked by reading, not run on hardware
//
#include <hip/hip_runtime.h>
#include <hip/hip_bf16.h>

// ---------------- problem constants ----------------
#define FEAT   64
#define INDIM  67           // feat(64) + xyz(3)
#define MSGDIM 134          // 2*INDIM
#define K1     160          // MSGDIM padded up to 5*32 for bf16 WMMA K=32
#define K1P    168          // LDS row stride (bf16) for K1 tiles, 16B-multiple
#define K2P    72           // LDS row stride (bf16) for 64-wide tiles, 16B-multiple
#define WAVES  4
#define TPB    128          // 4 wave32s
#define EDGES_PER_TILE (WAVES * 16)   // 64

typedef __attribute__((ext_vector_type(16))) __bf16 v16bf;
typedef __attribute__((ext_vector_type(8)))  __bf16 v8bf;
typedef __attribute__((ext_vector_type(2)))  __bf16 v2bf;
typedef __attribute__((ext_vector_type(8)))  float  v8f;

// order-preserving float<->int encoding for integer atomic max
__device__ __forceinline__ int enc_f32(float f) {
    int i = __float_as_int(f);
    return (i >= 0) ? i : (i ^ 0x7FFFFFFF);
}
__device__ __forceinline__ float dec_f32(int k) {
    int i = (k >= 0) ? k : (k ^ 0x7FFFFFFF);
    return __int_as_float(i);
}

// packed f32x2 -> bf16x2 store (lowers to v_cvt_pk_bf16_f32 + ds_store_b32)
__device__ __forceinline__ void st_pk(__bf16* p, float a, float b) {
    v2bf t; t[0] = (__bf16)a; t[1] = (__bf16)b;
    *(v2bf*)p = t;
}

// K-permutation: new message layout -> original W1 row index
//  new [0..63]   = x_i feat      -> old k
//  new [64..127] = d feat        -> old k+3   (old 67..130)
//  new [128..130]= x_i xyz       -> old k-64  (old 64..66)
//  new [131..133]= d xyz         -> old k     (old 131..133)
//  new [134..)   = zero pad
__device__ __forceinline__ int w1_old_row(int k) {
    if (k < 64)  return k;
    if (k < 128) return k + 3;
    if (k < 131) return k - 64;
    return k;  // 131..133
}

// Load one 16-bit A/B WMMA fragment chunk from LDS (row-major, K-contig).
// CDNA5 16-bit 16x32 layout: lane holds K=[koff..koff+7] in regs 0-3 and
// K=[16+koff..16+koff+7] in regs 4-7, koff = (lane>=16)*8.
__device__ __forceinline__ v16bf ld_frag(const __bf16* p, int koff) {
    v8bf lo = *(const v8bf*)(p + koff);
    v8bf hi = *(const v8bf*)(p + 16 + koff);
    return __builtin_shufflevector(lo, hi, 0, 1, 2, 3, 4, 5, 6, 7,
                                           8, 9, 10, 11, 12, 13, 14, 15);
}

__global__ void init_keys_kernel(int* __restrict__ keys, int n) {
    int i = blockIdx.x * blockDim.x + threadIdx.x;
    if (i < n) keys[i] = (int)0x80000000;   // INT_MIN sentinel == "no edge"
}

__global__ void finalize_keys_kernel(int* keys, float* out, int n) {
    int i = blockIdx.x * blockDim.x + threadIdx.x;
    if (i < n) {
        int k = keys[i];
        out[i] = (k == (int)0x80000000) ? 0.0f : dec_f32(k);   // deg==0 -> 0
    }
}

__global__ void __launch_bounds__(TPB)
edgeconv_wmma_kernel(const float* __restrict__ xyz,
                     const float* __restrict__ feat,
                     const int*   __restrict__ ei,     // [2, E] int32
                     const float* __restrict__ W1,     // [134, 64]
                     const float* __restrict__ b1,     // [64]
                     const float* __restrict__ W2,     // [64, 64]
                     const float* __restrict__ b2,     // [64]
                     int* __restrict__ keys,           // [N, 64] ordered-int
                     int E, int ntiles)
{
    __shared__ __align__(16) __bf16 WT1s[64 * K1P];           // permuted W1^T
    __shared__ __align__(16) __bf16 WT2s[64 * K2P];           // W2^T
    __shared__ __align__(16) __bf16 A1s[WAVES * 16 * K1P];    // per-wave message tile
    __shared__ __align__(16) __bf16 A2s[WAVES * 16 * K2P];    // per-wave hidden tile
    __shared__ int SDs[WAVES * 32];                           // per-wave [src16|dst16]

    const int tid  = threadIdx.x;
    const int wave = tid >> 5;
    const int lane = tid & 31;
    const int mrow = lane & 15;            // row (A) / col (B,C) index
    const int koff = (lane >> 4) * 8;      // K sub-offset; also C row base

    __bf16* A1 = A1s + wave * 16 * K1P;
    __bf16* A2 = A2s + wave * 16 * K2P;
    int*    SD = SDs + wave * 32;

    // ---- stage transposed (K-permuted) bf16 weights once per block ----
    for (int idx = tid; idx < 64 * K1P; idx += TPB) {
        int n = idx / K1P, k = idx - n * K1P;
        float w = (k < MSGDIM) ? W1[w1_old_row(k) * 64 + n] : 0.0f;
        WT1s[n * K1P + k] = (__bf16)w;
    }
    for (int idx = tid; idx < 64 * K2P; idx += TPB) {
        int n = idx / K2P, k = idx - n * K2P;
        float w = (k < 64) ? W2[k * 64 + n] : 0.0f;
        WT2s[n * K2P + k] = (__bf16)w;
    }
    __syncthreads();

    // biases are loop-invariant per lane: hoist
    float bb1[4], bb2[4];
#pragma unroll
    for (int nt = 0; nt < 4; ++nt) {
        bb1[nt] = b1[nt * 16 + mrow];
        bb2[nt] = b2[nt * 16 + mrow];
    }

    // per-lane LDS dest (byte address) and global dword index split for the
    // async edge-index load: lanes 0-15 fetch src[], lanes 16-31 fetch dst[]
    const unsigned sd_lds = (unsigned)(uintptr_t)(SD + lane);

    for (int t = blockIdx.x; t < ntiles; t += gridDim.x) {
        const int ebase = t * EDGES_PER_TILE + wave * 16;

        // ---- async edge-index tile load: global -> LDS, ASYNCcnt tracked ----
        {
            int ee = ebase + mrow;
            if (ee >= E) ee = E - 1;                    // clamp: keep EXEC full
            unsigned goff = (unsigned)(((lane < 16) ? ee : E + ee) * 4);
            // prior DS reads of SD must drain before the async write can land
            asm volatile("s_wait_dscnt 0\n\t"
                         "global_load_async_to_lds_b32 %0, %1, %2\n\t"
                         "s_wait_asynccnt 0"
                         :: "v"(sd_lds), "v"(goff), "s"(ei)
                         : "memory");
        }

        // ---- prefetch next tile's node feature rows (256B each) ----
        int tn = t + gridDim.x;
        if (tn < ntiles) {
            int e2 = tn * EDGES_PER_TILE + wave * 16 + mrow;
            if (e2 >= E) e2 = E - 1;
            int node = (lane < 16) ? ei[e2] : ei[E + e2];
            const float* rowp = feat + node * FEAT;
            __builtin_prefetch(rowp, 0, 1);
            __builtin_prefetch(rowp + 32, 0, 1);
        }

        // ---- gather: build 16 x K1 permuted message tile ----
        for (int r = 0; r < 16; ++r) {
            const int sj = SD[r];
            const int di = SD[16 + r];
            __bf16* row = A1 + r * K1P;
            // 128 feat columns: 2 per lane, one float2 load each side
            const float2 fi = *(const float2*)(feat + di * FEAT + 2 * lane);
            const float2 fj = *(const float2*)(feat + sj * FEAT + 2 * lane);
            st_pk(row + 2 * lane,       fi.x, fi.y);               // cols 0..63
            st_pk(row + 64 + 2 * lane,  fj.x - fi.x, fj.y - fi.y); // cols 64..127
            // tail cols 128..167: xyz_i(3), dxyz(3), zero pad; lanes 0..19
            if (lane < 20) {
                int c0 = 128 + 2 * lane;
                float a = 0.0f, b = 0.0f;
                if (c0 == 128) { a = xyz[di * 3 + 0]; b = xyz[di * 3 + 1]; }
                else if (c0 == 130) { a = xyz[di * 3 + 2];
                                      b = xyz[sj * 3 + 0] - xyz[di * 3 + 0]; }
                else if (c0 == 132) { a = xyz[sj * 3 + 1] - xyz[di * 3 + 1];
                                      b = xyz[sj * 3 + 2] - xyz[di * 3 + 2]; }
                st_pk(row + c0, a, b);
            }
        }
        __builtin_amdgcn_wave_barrier();

        // ---- GEMM1: (16 x 160) @ (160 x 64), bias + relu, restage bf16 ----
        const __bf16* a1row = A1 + mrow * K1P;
#pragma unroll
        for (int nt = 0; nt < 4; ++nt) {
            v8f acc = {};
            const __bf16* b1row = WT1s + (nt * 16 + mrow) * K1P;
#pragma unroll
            for (int kt = 0; kt < 5; ++kt) {
                v16bf a = ld_frag(a1row + kt * 32, koff);
                v16bf b = ld_frag(b1row + kt * 32, koff);
                acc = __builtin_amdgcn_wmma_f32_16x16x32_bf16(
                          false, a, false, b, (short)0, acc, false, false);
            }
#pragma unroll
            for (int v = 0; v < 8; ++v) {   // C layout: row = v + koff, col = mrow
                float h = fmaxf(acc[v] + bb1[nt], 0.0f);
                A2[(v + koff) * K2P + nt * 16 + mrow] = (__bf16)h;
            }
        }
        __builtin_amdgcn_wave_barrier();

        // ---- GEMM2: (16 x 64) @ (64 x 64), bias, atomic scatter-max ----
        // destination nodes for this lane's 8 C rows (shared across nt)
        int nodes[8];
#pragma unroll
        for (int v = 0; v < 8; ++v) nodes[v] = SD[16 + v + koff];

        const __bf16* a2row = A2 + mrow * K2P;
#pragma unroll
        for (int nt = 0; nt < 4; ++nt) {
            v8f acc = {};
            const __bf16* b2row = WT2s + (nt * 16 + mrow) * K2P;
#pragma unroll
            for (int kt = 0; kt < 2; ++kt) {
                v16bf a = ld_frag(a2row + kt * 32, koff);
                v16bf b = ld_frag(b2row + kt * 32, koff);
                acc = __builtin_amdgcn_wmma_f32_16x16x32_bf16(
                          false, a, false, b, (short)0, acc, false, false);
            }
#pragma unroll
            for (int v = 0; v < 8; ++v) {
                int e = ebase + (v + koff);
                if (e < E) {
                    atomicMax(&keys[nodes[v] * FEAT + nt * 16 + mrow],
                              enc_f32(acc[v] + bb2[nt]));
                }
            }
        }
    }
}

extern "C" void kernel_launch(void* const* d_in, const int* in_sizes, int n_in,
                              void* d_out, int out_size, void* d_ws, size_t ws_size,
                              hipStream_t stream) {
    const float* xyz  = (const float*)d_in[0];
    const float* feat = (const float*)d_in[1];
    const int*   ei   = (const int*)  d_in[2];   // JAX x64 off -> int32
    const float* W1   = (const float*)d_in[3];
    const float* b1   = (const float*)d_in[4];
    const float* W2   = (const float*)d_in[5];
    const float* b2   = (const float*)d_in[6];

    const int E = in_sizes[2] / 2;               // 1,600,000
    const int N = in_sizes[0] / 3;               // 100,000
    const int nelem = N * FEAT;

    int*   keys = (int*)d_out;                   // reuse d_out as key buffer
    float* out  = (float*)d_out;

    init_keys_kernel<<<(nelem + 255) / 256, 256, 0, stream>>>(keys, nelem);

    const int ntiles = (E + EDGES_PER_TILE - 1) / EDGES_PER_TILE;
    int grid = ntiles < 2048 ? ntiles : 2048;
    edgeconv_wmma_kernel<<<grid, TPB, 0, stream>>>(xyz, feat, ei, W1, b1, W2, b2,
                                                   keys, E, ntiles);

    finalize_keys_kernel<<<(nelem + 255) / 256, 256, 0, stream>>>(keys, out, nelem);
}
